// SequenceMemoryUpdater_58033598104188
// MI455X (gfx1250) — matrix operand order repaired and checked by reading, not compiled, not run on hardware
//
#include <hip/hip_runtime.h>
#include <hip/hip_bf16.h>

// Problem constants (fixed by the reference)
#define N_BATCH   20000
#define NNODES    50000
#define KMSG      8
#define DDIM      256
#define HDIM      256
#define PDEP      3

typedef __bf16 bf16_t;
typedef __bf16 v16bf __attribute__((ext_vector_type(16)));
typedef __bf16 v8bf  __attribute__((ext_vector_type(8)));
typedef float  v8f   __attribute__((ext_vector_type(8)));
typedef float  v4f   __attribute__((ext_vector_type(4)));

static __device__ __forceinline__ v8f wmma_bf16(v16bf a, v16bf b, v8f c) {
  // D = A(16x32 bf16) * B(32x16 bf16) + C(16x16 f32)
  return __builtin_amdgcn_wmma_f32_16x16x32_bf16(false, a, false, b, (short)0, c, false, false);
}

static __device__ __forceinline__ v8f vzero8() {
  v8f z;
  #pragma unroll
  for (int j = 0; j < 8; ++j) z[j] = 0.0f;
  return z;
}

static __device__ __forceinline__ float sigmoidf_(float x) {
  return 1.0f / (1.0f + __expf(-x));
}

// ---- Fragment loaders --------------------------------------------------
// A-fragment (16x32 bf16, MxK): lane (hi=lane>>4, m=lane&15) holds
//   A[m][d0+hi*8 + j]        for j=0..7   (frag elems 0..7)
//   A[m][d0+16+hi*8 + j]     for j=0..7   (frag elems 8..15)
static __device__ __forceinline__ v16bf load_a_from_f32(const float* __restrict__ row, int d0, int hi) {
  const float* p1 = row + d0 + hi * 8;
  const float* p2 = row + d0 + 16 + hi * 8;
  v4f x0 = *(const v4f*)(p1);
  v4f x1 = *(const v4f*)(p1 + 4);
  v4f x2 = *(const v4f*)(p2);
  v4f x3 = *(const v4f*)(p2 + 4);
  v16bf a;
  #pragma unroll
  for (int j = 0; j < 4; ++j) {
    a[j]      = (bf16_t)x0[j];
    a[4 + j]  = (bf16_t)x1[j];
    a[8 + j]  = (bf16_t)x2[j];
    a[12 + j] = (bf16_t)x3[j];
  }
  return a;
}

static __device__ __forceinline__ v16bf load_a_from_bf16(const bf16_t* __restrict__ row, int d0, int hi) {
  v8bf lo = *(const v8bf*)(row + d0 + hi * 8);
  v8bf hc = *(const v8bf*)(row + d0 + 16 + hi * 8);
  v16bf a;
  #pragma unroll
  for (int j = 0; j < 8; ++j) { a[j] = lo[j]; a[8 + j] = hc[j]; }
  return a;
}

// B-fragment (32x16 bf16, KxN) for C[m,e]=sum_d A[m,d]*W[e,d]:
//   B[k][n] = W[e0+n][d0+k]; lane (hi,n) holds k = hi*8 + 0..7 and
//   16 + hi*8 + 0..7  -> two 16B loads from W row (e0+n).
static __device__ __forceinline__ v16bf load_b_frag(const bf16_t* __restrict__ W, int ldw,
                                                    int e0, int d0, int lane) {
  const int n = lane & 15, hi = lane >> 4;
  const bf16_t* r = W + (size_t)(e0 + n) * ldw + d0 + hi * 8;
  v8bf lo = *(const v8bf*)(r);
  v8bf hc = *(const v8bf*)(r + 16);
  v16bf b;
  #pragma unroll
  for (int j = 0; j < 8; ++j) { b[j] = lo[j]; b[8 + j] = hc[j]; }
  return b;
}

// ---- Kernels -----------------------------------------------------------

__global__ void cvt_f32_bf16_kernel(const float* __restrict__ s, bf16_t* __restrict__ d, int n) {
  int i = blockIdx.x * blockDim.x + threadIdx.x;
  const int stride = gridDim.x * blockDim.x;
  for (; i < n; i += stride) d[i] = (bf16_t)s[i];
}

// Pass i: pool[n][e] = max_k relu( sum_d other[ids[n]][k][d] * W[e][d] + b[e] )
// One block = 2 nodes (16 message rows = one WMMA M-tile); 4 waves x 4 e-tiles.
__global__ void __launch_bounds__(128) fc1_pool_kernel(
    const float* __restrict__ omsg_p,   // [NNODES][KMSG][DDIM] (this pass)
    const int* __restrict__ ids,        // [N_BATCH]
    const bf16_t* __restrict__ w,       // [DDIM][DDIM] bf16
    const float* __restrict__ bias,     // [DDIM]
    bf16_t* __restrict__ pool)          // [N_BATCH][DDIM]
{
  const int tile = blockIdx.x;          // node pair
  const int lane = threadIdx.x & 31;
  const int wave = threadIdx.x >> 5;
  const int hi   = lane >> 4;
  const int m    = lane & 15;           // A row: node = tile*2 + (m>>3), msg k = m&7

  const int nodeA = tile * 2 + (m >> 3);
  const float* arow = omsg_p + ((size_t)ids[nodeA] * KMSG + (m & 7)) * DDIM;

  v8f acc[4];
  #pragma unroll
  for (int t = 0; t < 4; ++t) acc[t] = vzero8();

  const int e_base = wave * 64;
  for (int d0 = 0; d0 < DDIM; d0 += 32) {
    v16bf a = load_a_from_f32(arow, d0, hi);
    #pragma unroll
    for (int t = 0; t < 4; ++t) {
      v16bf b = load_b_frag(w, DDIM, e_base + t * 16, d0, lane);
      acc[t] = wmma_bf16(a, b, acc[t]);
    }
  }

  // C-frag: lane column = lane&15; rows r+8*hi -> this lane owns exactly the
  // 8 messages of node (tile*2 + hi): max-pool is a per-lane reduction.
  const int col  = lane & 15;
  const int node = tile * 2 + hi;
  #pragma unroll
  for (int t = 0; t < 4; ++t) {
    const int e = e_base + t * 16 + col;
    const float bv = bias[e];
    float mx = 0.0f;                    // relu folds into max with 0
    #pragma unroll
    for (int r = 0; r < 8; ++r) mx = fmaxf(mx, acc[t][r] + bv);
    pool[(size_t)node * DDIM + e] = (bf16_t)mx;
  }
}

// pre_out = [pre_in | pool] @ W^T + b   (K = 512)
__global__ void __launch_bounds__(128) fc2_kernel(
    const bf16_t* __restrict__ pre_in,  // [N_BATCH][DDIM]
    const bf16_t* __restrict__ pool,    // [N_BATCH][DDIM]
    const bf16_t* __restrict__ w,       // [DDIM][2*DDIM] bf16
    const float* __restrict__ bias,     // [DDIM]
    bf16_t* __restrict__ pre_out)       // [N_BATCH][DDIM]
{
  const int tile = blockIdx.x;          // 16 nodes
  const int lane = threadIdx.x & 31;
  const int wave = threadIdx.x >> 5;
  const int hi   = lane >> 4;
  const int m    = lane & 15;
  const int node = tile * 16 + m;

  const bf16_t* prow = pre_in + (size_t)node * DDIM;
  const bf16_t* qrow = pool   + (size_t)node * DDIM;

  v8f acc[4];
  #pragma unroll
  for (int t = 0; t < 4; ++t) acc[t] = vzero8();

  const int e_base = wave * 64;
  for (int d0 = 0; d0 < 2 * DDIM; d0 += 32) {
    v16bf a = (d0 < DDIM) ? load_a_from_bf16(prow, d0, hi)
                          : load_a_from_bf16(qrow, d0 - DDIM, hi);
    #pragma unroll
    for (int t = 0; t < 4; ++t) {
      v16bf b = load_b_frag(w, 2 * DDIM, e_base + t * 16, d0, lane);
      acc[t] = wmma_bf16(a, b, acc[t]);
    }
  }

  const int col = lane & 15;
  #pragma unroll
  for (int t = 0; t < 4; ++t) {
    const int e = e_base + t * 16 + col;
    const float bv = bias[e];
    #pragma unroll
    for (int r = 0; r < 8; ++r) {
      const int nr = tile * 16 + hi * 8 + r;
      pre_out[(size_t)nr * DDIM + e] = (bf16_t)(acc[t][r] + bv);
    }
  }
}

// GRU cell: 6 accumulator chains (i_r,i_z,i_n,h_r,h_z,h_n) per 16x16 tile,
// then gates locally (bias uniform per lane since column is fixed per lane).
__global__ void __launch_bounds__(128) gru_kernel(
    const bf16_t* __restrict__ pre,     // [N_BATCH][DDIM]
    const float* __restrict__ memory,   // [NNODES][HDIM]
    const int* __restrict__ ids,        // [N_BATCH]
    const bf16_t* __restrict__ wih,     // [3H][DDIM] bf16
    const bf16_t* __restrict__ whh,     // [3H][HDIM] bf16
    const float* __restrict__ bih,      // [3H]
    const float* __restrict__ bhh,      // [3H]
    float* __restrict__ newh)           // [N_BATCH][HDIM]
{
  const int tile = blockIdx.x;          // 16 nodes
  const int lane = threadIdx.x & 31;
  const int wave = threadIdx.x >> 5;
  const int hi   = lane >> 4;
  const int m    = lane & 15;
  const int node = tile * 16 + m;

  const bf16_t* prow = pre + (size_t)node * DDIM;
  const float*  hrow = memory + (size_t)ids[node] * HDIM;

  for (int ct = 0; ct < 4; ++ct) {
    const int c0 = (wave * 4 + ct) * 16;
    v8f ir = vzero8(), iz = vzero8(), inn = vzero8();
    v8f hr = vzero8(), hz = vzero8(), hn  = vzero8();

    for (int d0 = 0; d0 < DDIM; d0 += 32) {
      v16bf ap = load_a_from_bf16(prow, d0, hi);
      v16bf ah = load_a_from_f32(hrow, d0, hi);
      ir  = wmma_bf16(ap, load_b_frag(wih, DDIM, c0,            d0, lane), ir);
      iz  = wmma_bf16(ap, load_b_frag(wih, DDIM, HDIM + c0,     d0, lane), iz);
      inn = wmma_bf16(ap, load_b_frag(wih, DDIM, 2 * HDIM + c0, d0, lane), inn);
      hr  = wmma_bf16(ah, load_b_frag(whh, HDIM, c0,            d0, lane), hr);
      hz  = wmma_bf16(ah, load_b_frag(whh, HDIM, HDIM + c0,     d0, lane), hz);
      hn  = wmma_bf16(ah, load_b_frag(whh, HDIM, 2 * HDIM + c0, d0, lane), hn);
    }

    const int col = lane & 15;
    const int e = c0 + col;
    const float bir = bih[e], biz = bih[HDIM + e], bin = bih[2 * HDIM + e];
    const float bhr = bhh[e], bhz = bhh[HDIM + e], bhn = bhh[2 * HDIM + e];
    #pragma unroll
    for (int r = 0; r < 8; ++r) {
      const int nr = tile * 16 + hi * 8 + r;
      const float hv = memory[(size_t)ids[nr] * HDIM + e];
      const float rg = sigmoidf_(ir[r] + bir + hr[r] + bhr);
      const float zg = sigmoidf_(iz[r] + biz + hz[r] + bhz);
      const float ng = tanhf(inn[r] + bin + rg * (hn[r] + bhn));
      newh[(size_t)nr * HDIM + e] = (1.0f - zg) * ng + zg * hv;
    }
  }
}

// Deterministic duplicate-id resolution: highest n wins (== last occurrence).
__global__ void winner_init_kernel(int* __restrict__ winner, int n) {
  int i = blockIdx.x * blockDim.x + threadIdx.x;
  if (i < n) winner[i] = -1;
}
__global__ void winner_mark_kernel(const int* __restrict__ ids, int* __restrict__ winner, int n) {
  int i = blockIdx.x * blockDim.x + threadIdx.x;
  if (i < n) atomicMax(&winner[ids[i]], i);
}
__global__ void scatter_kernel(const int* __restrict__ ids, const int* __restrict__ winner,
                               const float* __restrict__ newh, float* __restrict__ out, int n) {
  const int idx = blockIdx.x * blockDim.x + threadIdx.x;
  if (idx >= n * HDIM) return;
  const int node = idx / HDIM;
  const int c    = idx - node * HDIM;
  const int row  = ids[node];
  if (winner[row] == node)
    out[(size_t)row * HDIM + c] = newh[(size_t)node * HDIM + c];
}

// ---- Launch ------------------------------------------------------------

extern "C" void kernel_launch(void* const* d_in, const int* in_sizes, int n_in,
                              void* d_out, int out_size, void* d_ws, size_t ws_size,
                              hipStream_t stream) {
  (void)in_sizes; (void)n_in; (void)out_size; (void)ws_size;
  const int*   ids    = (const int*)  d_in[0];
  const float* umsg   = (const float*)d_in[1];
  const float* omsg   = (const float*)d_in[2];
  const float* memory = (const float*)d_in[3];
  const float* fc1_w  = (const float*)d_in[4];
  const float* fc1_b  = (const float*)d_in[5];
  const float* fc2_w  = (const float*)d_in[6];
  const float* fc2_b  = (const float*)d_in[7];
  const float* w_ih   = (const float*)d_in[8];
  const float* w_hh   = (const float*)d_in[9];
  const float* b_ih   = (const float*)d_in[10];
  const float* b_hh   = (const float*)d_in[11];
  float* out = (float*)d_out;

  // Workspace layout
  char*  ws  = (char*)d_ws;
  size_t off = 0;
  auto carve = [&](size_t bytes) -> void* {
    void* p = ws + off;
    off += (bytes + 255) & ~(size_t)255;
    return p;
  };
  bf16_t* fc1w_b = (bf16_t*)carve((size_t)PDEP * DDIM * DDIM * sizeof(bf16_t));
  bf16_t* fc2w_b = (bf16_t*)carve((size_t)PDEP * DDIM * 2 * DDIM * sizeof(bf16_t));
  bf16_t* wih_b  = (bf16_t*)carve((size_t)3 * HDIM * DDIM * sizeof(bf16_t));
  bf16_t* whh_b  = (bf16_t*)carve((size_t)3 * HDIM * HDIM * sizeof(bf16_t));
  bf16_t* preA   = (bf16_t*)carve((size_t)N_BATCH * DDIM * sizeof(bf16_t));
  bf16_t* preB   = (bf16_t*)carve((size_t)N_BATCH * DDIM * sizeof(bf16_t));
  bf16_t* pool   = (bf16_t*)carve((size_t)N_BATCH * DDIM * sizeof(bf16_t));
  float*  newh   = (float*) carve((size_t)N_BATCH * HDIM * sizeof(float));
  int*    winner = (int*)   carve((size_t)NNODES * sizeof(int));

  // 1) Output starts as a copy of memory (untouched rows pass through).
  hipMemcpyAsync(out, memory, (size_t)NNODES * HDIM * sizeof(float),
                 hipMemcpyDeviceToDevice, stream);

  // 2) One-shot bf16 conversions (weights + initial pre = unique_messages).
  const int CVT_T = 256, CVT_B = 1024;
  cvt_f32_bf16_kernel<<<CVT_B, CVT_T, 0, stream>>>(fc1_w, fc1w_b, PDEP * DDIM * DDIM);
  cvt_f32_bf16_kernel<<<CVT_B, CVT_T, 0, stream>>>(fc2_w, fc2w_b, PDEP * DDIM * 2 * DDIM);
  cvt_f32_bf16_kernel<<<CVT_B, CVT_T, 0, stream>>>(w_ih,  wih_b,  3 * HDIM * DDIM);
  cvt_f32_bf16_kernel<<<CVT_B, CVT_T, 0, stream>>>(w_hh,  whh_b,  3 * HDIM * HDIM);
  cvt_f32_bf16_kernel<<<CVT_B, CVT_T, 0, stream>>>(umsg,  preA,   N_BATCH * DDIM);

  // 3) Three message-passing stages.
  const bf16_t* pre_cur = preA;
  bf16_t*       pre_nxt = preB;
  for (int p = 0; p < PDEP; ++p) {
    fc1_pool_kernel<<<N_BATCH / 2, 128, 0, stream>>>(
        omsg + (size_t)p * NNODES * KMSG * DDIM, ids,
        fc1w_b + (size_t)p * DDIM * DDIM, fc1_b + (size_t)p * DDIM, pool);
    fc2_kernel<<<N_BATCH / 16, 128, 0, stream>>>(
        pre_cur, pool, fc2w_b + (size_t)p * DDIM * 2 * DDIM,
        fc2_b + (size_t)p * DDIM, pre_nxt);
    const bf16_t* t = pre_cur; pre_cur = pre_nxt; pre_nxt = (bf16_t*)t;
  }

  // 4) GRU memory update.
  gru_kernel<<<N_BATCH / 16, 128, 0, stream>>>(
      pre_cur, memory, ids, wih_b, whh_b, b_ih, b_hh, newh);

  // 5) Deterministic scatter (last occurrence of a duplicate id wins).
  winner_init_kernel<<<(NNODES + 255) / 256, 256, 0, stream>>>(winner, NNODES);
  winner_mark_kernel<<<(N_BATCH + 255) / 256, 256, 0, stream>>>(ids, winner, N_BATCH);
  scatter_kernel<<<(N_BATCH * HDIM + 255) / 256, 256, 0, stream>>>(ids, winner, newh, out, N_BATCH);
}
